// Cif_66984309948685
// MI455X (gfx1250) — compile-verified, matrix operand order checked by reading, not run on hardware
//
#include <hip/hip_runtime.h>
#include <cmath>

// ---------------------------------------------------------------------------
// CIF conformer conv-module + integrate-and-fire for MI455X (gfx1250, wave32)
// GEMMs: bf16 WMMA (v_wmma_f32_16x16x32_bf16), LDS tiles staged via
// async global->LDS loads and the Tensor Data Mover where the toolchain
// exposes them (guarded; falls back to the sync path otherwise).
// ---------------------------------------------------------------------------

typedef __attribute__((ext_vector_type(16))) __bf16 v16bf;
typedef __attribute__((ext_vector_type(8)))  __bf16 v8bf;
typedef __attribute__((ext_vector_type(8)))  float  v8f;

typedef unsigned int u32x4 __attribute__((ext_vector_type(4)));
typedef int          i32x4 __attribute__((ext_vector_type(4)));
typedef int          i32x8 __attribute__((ext_vector_type(8)));

#define AS1 __attribute__((address_space(1)))
#define AS3 __attribute__((address_space(3)))

#if defined(__has_builtin)
#if __has_builtin(__builtin_amdgcn_global_load_async_to_lds_b128)
#define CIF_HAS_ASYNC 1
#endif
#if __has_builtin(__builtin_amdgcn_tensor_load_to_lds)
#define CIF_HAS_TDM 1
#endif
#endif

static constexpr int   B_      = 16;
static constexpr int   T_      = 2048;
static constexpr int   C_      = 256;
static constexpr int   MTOT    = B_ * T_;      // 32768 rows
static constexpr int   CMAX    = T_ + 1;       // 2049
static constexpr float TH_     = 1.0f;
static constexpr float PADV_   = -1.0f;
static constexpr float EPS_    = 1e-5f;

__device__ __forceinline__ float sigmoidf_(float x) {
  return 1.0f / (1.0f + __expf(-x));
}

// 16-byte global -> LDS stage. Async (writes LDS directly, ASYNCcnt) when the
// toolchain has the gfx1250 builtin, else synchronous VGPR round-trip.
// Builtin signature (from the round-2 diagnostic): AS1 v4i* src, AS3 v4i* dst,
// imm offset, imm cpol.
__device__ __forceinline__ void stage16B(const __bf16* g, __bf16* l) {
#ifdef CIF_HAS_ASYNC
  __builtin_amdgcn_global_load_async_to_lds_b128((AS1 i32x4*)g, (AS3 i32x4*)l, 0, 0);
#else
  *reinterpret_cast<v8bf*>(l) = *reinterpret_cast<const v8bf*>(g);
#endif
}

__device__ __forceinline__ void wait_async_stages() {
#ifdef CIF_HAS_ASYNC
#if __has_builtin(__builtin_amdgcn_s_wait_asynccnt)
  __builtin_amdgcn_s_wait_asynccnt(0);
#else
  asm volatile("s_wait_asynccnt 0x0" ::: "memory");
#endif
#endif
}

#ifdef CIF_HAS_TDM
__device__ __forceinline__ unsigned lds_byte_off(const void* p) {
  return (unsigned)(size_t)(AS3 const void*)p;   // generic -> LDS offset
}

// Issue one TDM 2D tile load: rows x cols bf16 tile from a row-major tensor
// (row stride `stride` elements, dim0 extent = stride here) into a packed LDS
// tile (row stride == cols). D# bit layout per CDNA5 ISA ch.8
// (count=1, type=2, data_size=2B).
__device__ __forceinline__ void tdm_load_tile_bf16(const __bf16* gbase,
                                                   unsigned lds_off,
                                                   unsigned rows, unsigned cols,
                                                   unsigned stride,
                                                   unsigned tensor_rows) {
  unsigned long long ga = (unsigned long long)gbase;
  u32x4 g0;
  g0[0] = 1u;                                        // count=1, user mode
  g0[1] = lds_off;                                   // lds_addr
  g0[2] = (unsigned)ga;                              // global_addr[31:0]
  g0[3] = (unsigned)((ga >> 32) & 0x1FFFFFFu) | (2u << 30);  // addr[56:32], type=2
  i32x8 g1;
  g1[0] = (int)(1u << 16);                           // data_size=1 (2 bytes)
  g1[1] = (int)((stride & 0xFFFFu) << 16);           // tensor_dim0[15:0] (K extent)
  g1[2] = (int)(((stride >> 16) & 0xFFFFu) | ((tensor_rows & 0xFFFFu) << 16)); // dim0 hi | tensor_dim1 lo
  g1[3] = (int)(((tensor_rows >> 16) & 0xFFFFu) | ((cols & 0xFFFFu) << 16));   // dim1 hi | tile_dim0
  g1[4] = (int)(rows & 0xFFFFu);                     // tile_dim1 | tile_dim2=0
  g1[5] = (int)stride;                               // tensor_dim0_stride[31:0]
  g1[6] = 0;                                         // stride0 hi | stride1 lo
  g1[7] = 0;                                         // stride1 hi
  i32x4 z4 = {0, 0, 0, 0};
#if __clang_major__ >= 23
  i32x8 z8 = {0, 0, 0, 0, 0, 0, 0, 0};
  __builtin_amdgcn_tensor_load_to_lds(g0, g1, z4, z4, z8, 0);
#else
  __builtin_amdgcn_tensor_load_to_lds(g0, g1, z4, z4, 0);
#endif
}
#endif  // CIF_HAS_TDM

// Build a 16-element bf16 WMMA fragment from an LDS tile (row-major, `stride`
// bf16 elements per row, 32-wide K panel). Layout per CDNA5 ISA 7.12.2:
//   lane L holds row (L&15); K base = (L>>4)*8; element e -> K = (e>>3)*16 + kb + (e&7)
// The two 8-element runs are K-contiguous -> two ds_load_b128 per fragment.
__device__ __forceinline__ v16bf frag_ld(const __bf16* base, int row, int stride) {
  const int lane = threadIdx.x & 31;
  const __bf16* p = base + (row + (lane & 15)) * stride + ((lane >> 4) << 3);
  v8bf lo = *reinterpret_cast<const v8bf*>(p);
  v8bf hi = *reinterpret_cast<const v8bf*>(p + 16);
  return __builtin_shufflevector(lo, hi, 0,1,2,3,4,5,6,7,8,9,10,11,12,13,14,15);
}

// ---------------------------------------------------------------------------
__global__ void k_cvt_bf16(const float* __restrict__ in, __bf16* __restrict__ out, int n) {
  int i = blockIdx.x * 256 + threadIdx.x;
  if (i < n) out[i] = (__bf16)in[i];
}

__global__ void k_init(float* __restrict__ alpha_acc, float* __restrict__ loss) {
  int i = blockIdx.x * 256 + threadIdx.x;
  if (i < MTOT) alpha_acc[i] = 0.0f;
  if (i == 0)  *loss = 0.0f;
}

// ---------------------------------------------------------------------------
// GEMM1 + GLU:  Y1[m, n] = (X@W1a^T + b1a) * sigmoid(X@W1g^T + b1g)
// Block tile: 64 rows x 64 output cols (128 accumulation cols: n and n+256).
// 8 waves = 4 (M) x 2 (N); each wave: 16 rows x 32 cols x {a, gate}.
// ---------------------------------------------------------------------------
__global__ __launch_bounds__(256)
void k_gemm1_glu(const __bf16* __restrict__ Xbf, const __bf16* __restrict__ W1bf,
                 const float* __restrict__ pw1_b, float* __restrict__ Y1) {
  __shared__ __bf16 ldsA[64 * 32];
  __shared__ __bf16 ldsW[128 * 32];

  const int n0   = blockIdx.x * 64;       // output-channel tile (0..255)
  const int m0   = blockIdx.y * 64;       // row tile
  const int tid  = threadIdx.x;
  const int lane = tid & 31;
  const int wave = tid >> 5;
  const int wm   = wave & 3;              // 0..3 : 16-row subtile
  const int wn   = wave >> 2;             // 0..1 : 32-col subtile

  v8f accA0 = {}, accA1 = {}, accG0 = {}, accG1 = {};

  const int ar = tid >> 2, ac = (tid & 3) << 3;      // A stage: 8 bf16/thread
  const int wr = tid >> 1, wc = (tid & 1) << 4;      // W stage: 16 bf16/thread
  const int wrow = (wr < 64) ? (n0 + wr) : (256 + n0 + (wr - 64));

  for (int k0 = 0; k0 < 256; k0 += 32) {
    // stage A tile (64x32) and W tile (128x32: 'a' rows then gate rows)
    stage16B(Xbf + (size_t)(m0 + ar) * 256 + k0 + ac, &ldsA[ar * 32 + ac]);
    stage16B(W1bf + (size_t)wrow * 256 + k0 + wc,     &ldsW[wr * 32 + wc]);
    stage16B(W1bf + (size_t)wrow * 256 + k0 + wc + 8, &ldsW[wr * 32 + wc + 8]);
    if (k0 + 32 < 256) {  // gfx1250 global_prefetch_b8 for next K panel
      __builtin_prefetch(Xbf + (size_t)(m0 + ar) * 256 + k0 + 32 + ac, 0, 1);
      __builtin_prefetch(W1bf + (size_t)wrow * 256 + k0 + 32 + wc, 0, 1);
    }
    wait_async_stages();
    __syncthreads();

    v16bf a   = frag_ld(ldsA, wm * 16, 32);
    v16bf bA0 = frag_ld(ldsW, wn * 32,            32);
    v16bf bA1 = frag_ld(ldsW, wn * 32 + 16,       32);
    v16bf bG0 = frag_ld(ldsW, 64 + wn * 32,       32);
    v16bf bG1 = frag_ld(ldsW, 64 + wn * 32 + 16,  32);

    accA0 = __builtin_amdgcn_wmma_f32_16x16x32_bf16(false, a, false, bA0, (short)0, accA0, false, false);
    accA1 = __builtin_amdgcn_wmma_f32_16x16x32_bf16(false, a, false, bA1, (short)0, accA1, false, false);
    accG0 = __builtin_amdgcn_wmma_f32_16x16x32_bf16(false, a, false, bG0, (short)0, accG0, false, false);
    accG1 = __builtin_amdgcn_wmma_f32_16x16x32_bf16(false, a, false, bG1, (short)0, accG1, false, false);
    __syncthreads();
  }

  // Epilogue: GLU.  C/D layout: reg r -> row m = 8*(lane>>4)+r, col = lane&15.
  const int mrow = m0 + wm * 16 + ((lane >> 4) << 3);
  const int nA   = n0 + wn * 32 + (lane & 15);
#pragma unroll
  for (int r = 0; r < 8; ++r) {
    const int m = mrow + r;
    {
      float av = accA0[r] + pw1_b[nA];
      float gv = accG0[r] + pw1_b[nA + 256];
      Y1[(size_t)m * C_ + nA] = av * sigmoidf_(gv);
    }
    {
      float av = accA1[r] + pw1_b[nA + 16];
      float gv = accG1[r] + pw1_b[nA + 16 + 256];
      Y1[(size_t)m * C_ + nA + 16] = av * sigmoidf_(gv);
    }
  }
}

// ---------------------------------------------------------------------------
// Depthwise conv-5 (SAME, along T) + BN1 + Swish, emits bf16 for GEMM2.
// ---------------------------------------------------------------------------
__global__ void k_dwconv(const float* __restrict__ Y1,
                         const float* __restrict__ dw_w, const float* __restrict__ dw_b,
                         const float* __restrict__ g, const float* __restrict__ bb,
                         const float* __restrict__ mm, const float* __restrict__ vv,
                         __bf16* __restrict__ Zbf) {
  int idx = blockIdx.x * 256 + threadIdx.x;
  if (idx >= MTOT * C_) return;
  const int c  = idx & (C_ - 1);
  const int bt = idx >> 8;              // b*T + t
  const int t  = bt & (T_ - 1);
  float z = dw_b[c];
#pragma unroll
  for (int k = 0; k < 5; ++k) {
    int tt = t + k - 2;
    if (tt >= 0 && tt < T_)
      z += Y1[(size_t)(bt + k - 2) * C_ + c] * dw_w[c * 5 + k];
  }
  z = (z - mm[c]) * rsqrtf(vv[c] + EPS_) * g[c] + bb[c];
  z = z * sigmoidf_(z);                 // Swish
  Zbf[idx] = (__bf16)z;
}

// ---------------------------------------------------------------------------
// GEMM2 + BN2 + ReLU + partial alpha dot: alpha_acc[m] += relu(bn2(o)) * lin_w[n]
// A tile staged by the Tensor Data Mover (TENSORcnt) when available.
// ---------------------------------------------------------------------------
__global__ __launch_bounds__(256)
void k_gemm2_fused(const __bf16* __restrict__ Zbf, const __bf16* __restrict__ W2bf,
                   const float* __restrict__ pw2_b,
                   const float* __restrict__ bn2_g, const float* __restrict__ bn2_b,
                   const float* __restrict__ bn2_m, const float* __restrict__ bn2_v,
                   const float* __restrict__ lin_w, float* __restrict__ alpha_acc) {
  __shared__ __bf16 ldsA[64 * 32];
  __shared__ __bf16 ldsW[64 * 32];

  const int n0   = blockIdx.x * 64;
  const int m0   = blockIdx.y * 64;
  const int tid  = threadIdx.x;
  const int lane = tid & 31;
  const int wave = tid >> 5;
  const int wm   = wave & 3;
  const int wn   = wave >> 2;

  v8f acc0 = {}, acc1 = {};
  const int ar = tid >> 2, ac = (tid & 3) << 3;

  for (int k0 = 0; k0 < 256; k0 += 32) {
#ifdef CIF_HAS_TDM
    // one TDM descriptor moves the whole 64x32 A tile; issue from wave 0 only
    if (wave == 0) {
      tdm_load_tile_bf16(Zbf + (size_t)m0 * 256 + k0, lds_byte_off(ldsA),
                         /*rows=*/64, /*cols=*/32, /*stride=*/256,
                         /*tensor_rows=*/(unsigned)MTOT);
    }
#else
    stage16B(Zbf + (size_t)(m0 + ar) * 256 + k0 + ac, &ldsA[ar * 32 + ac]);
#endif
    stage16B(W2bf + (size_t)(n0 + ar) * 256 + k0 + ac, &ldsW[ar * 32 + ac]);
    if (k0 + 32 < 256) {
      __builtin_prefetch(Zbf + (size_t)(m0 + ar) * 256 + k0 + 32 + ac, 0, 1);
      __builtin_prefetch(W2bf + (size_t)(n0 + ar) * 256 + k0 + 32 + ac, 0, 1);
    }
    wait_async_stages();
#ifdef CIF_HAS_TDM
    __builtin_amdgcn_s_wait_tensorcnt(0);   // SALU: harmless for non-issuing waves
#endif
    __syncthreads();

    v16bf a  = frag_ld(ldsA, wm * 16, 32);
    v16bf b0 = frag_ld(ldsW, wn * 32,      32);
    v16bf b1 = frag_ld(ldsW, wn * 32 + 16, 32);
    acc0 = __builtin_amdgcn_wmma_f32_16x16x32_bf16(false, a, false, b0, (short)0, acc0, false, false);
    acc1 = __builtin_amdgcn_wmma_f32_16x16x32_bf16(false, a, false, b1, (short)0, acc1, false, false);
    __syncthreads();
  }

  const int mrow = m0 + wm * 16 + ((lane >> 4) << 3);
#pragma unroll
  for (int r = 0; r < 8; ++r) {
    const int m = mrow + r;
#pragma unroll
    for (int j = 0; j < 2; ++j) {
      const int n = n0 + wn * 32 + j * 16 + (lane & 15);
      float o = (j ? acc1[r] : acc0[r]) + pw2_b[n];
      o = (o - bn2_m[n]) * rsqrtf(bn2_v[n] + EPS_) * bn2_g[n] + bn2_b[n];
      o = fmaxf(o, 0.0f);
      atomicAdd(&alpha_acc[m], o * lin_w[n]);
    }
  }
}

// ---------------------------------------------------------------------------
// alpha = mask ? sigmoid(acc + lin_b) : 0 ; loss_pen = sum(alpha)
// ---------------------------------------------------------------------------
__global__ void k_alpha(const float* __restrict__ alpha_acc,
                        const unsigned char* __restrict__ hs_mask,
                        const float* __restrict__ lin_b,
                        float* __restrict__ alpha, float* __restrict__ loss) {
  const int i = blockIdx.x * 256 + threadIdx.x;
  float a = 0.0f;
  if (i < MTOT) {
    a = sigmoidf_(alpha_acc[i] + lin_b[0]);
    if (!hs_mask[i]) a = 0.0f;
    alpha[i] = a;
  }
  __shared__ float red[256];
  red[threadIdx.x] = a;
  __syncthreads();
  for (int s = 128; s > 0; s >>= 1) {
    if (threadIdx.x < s) red[threadIdx.x] += red[threadIdx.x + s];
    __syncthreads();
  }
  if (threadIdx.x == 0) atomicAdd(loss, red[0]);
}

// ---------------------------------------------------------------------------
// CIF integrate-and-fire: one block per sequence, one lane per channel.
// acc / fire decisions are uniform across the block (all lanes redo them).
// ---------------------------------------------------------------------------
__global__ __launch_bounds__(256)
void k_cif(const float* __restrict__ alpha, const float* __restrict__ hs,
           float* __restrict__ cs, int* __restrict__ count) {
  const int b = blockIdx.x;
  const int c = threadIdx.x;
  const float* al  = alpha + (size_t)b * T_;
  const float* h   = hs + (size_t)b * T_ * C_ + c;
  float* out       = cs + (size_t)b * CMAX * C_ + c;

  float acc = 0.0f, hacc = 0.0f;
  int n = 0;
  for (int t = 0; t < T_; ++t) {
    const float a  = al[t];
    const float hu = h[(size_t)t * C_];
    const float acc2 = acc + a;
    if (acc2 >= TH_ - 0.1f) {                       // fire
      out[(size_t)n * C_] = hacc + (TH_ - acc) * hu;
      ++n;
      hacc = (acc2 - TH_) * hu;
      acc  = acc2 - TH_;
    } else {
      hacc += a * hu;
      acc   = acc2;
    }
  }
  if (acc >= TH_ * 0.5f) {                          // tail fire
    out[(size_t)n * C_] = hacc;
    ++n;
  }
  for (int i = n; i < CMAX; ++i) out[(size_t)i * C_] = PADV_;
  if (c == 0) count[b] = n;
}

// ---------------------------------------------------------------------------
// cs_mask[b,i,j] = (i < count[b]) && (j < count[b])
// ---------------------------------------------------------------------------
__global__ void k_mask(const int* __restrict__ count, float* __restrict__ mask_out) {
  const size_t idx = (size_t)blockIdx.x * 256 + threadIdx.x;
  const size_t per = (size_t)CMAX * CMAX;
  if (idx >= (size_t)B_ * per) return;
  const int b = (int)(idx / per);
  const size_t rem = idx % per;
  const int i = (int)(rem / CMAX);
  const int j = (int)(rem % CMAX);
  const int cnt = count[b];
  mask_out[idx] = (i < cnt && j < cnt) ? 1.0f : 0.0f;
}

// ---------------------------------------------------------------------------
extern "C" void kernel_launch(void* const* d_in, const int* in_sizes, int n_in,
                              void* d_out, int out_size, void* d_ws, size_t ws_size,
                              hipStream_t stream) {
  const float*         hs_pad  = (const float*)d_in[0];
  const unsigned char* hs_mask = (const unsigned char*)d_in[1];
  const float* pw1_w = (const float*)d_in[2];
  const float* pw1_b = (const float*)d_in[3];
  const float* dw_w  = (const float*)d_in[4];
  const float* dw_b  = (const float*)d_in[5];
  const float* bn1_g = (const float*)d_in[6];
  const float* bn1_b = (const float*)d_in[7];
  const float* bn1_m = (const float*)d_in[8];
  const float* bn1_v = (const float*)d_in[9];
  const float* pw2_w = (const float*)d_in[10];
  const float* pw2_b = (const float*)d_in[11];
  const float* bn2_g = (const float*)d_in[12];
  const float* bn2_b = (const float*)d_in[13];
  const float* bn2_m = (const float*)d_in[14];
  const float* bn2_v = (const float*)d_in[15];
  const float* lin_w = (const float*)d_in[16];
  const float* lin_b = (const float*)d_in[17];

  // workspace carve-up (all offsets 256B-aligned)
  char* ws = (char*)d_ws;
  __bf16* Xbf       = (__bf16*)ws;                 ws += (size_t)MTOT * C_ * 2;   // 16 MB
  __bf16* W1bf      = (__bf16*)ws;                 ws += (size_t)512 * 256 * 2;
  __bf16* W2bf      = (__bf16*)ws;                 ws += (size_t)256 * 256 * 2;
  float*  Y1        = (float*)ws;                  ws += (size_t)MTOT * C_ * 4;   // 32 MB
  __bf16* Zbf       = (__bf16*)ws;                 ws += (size_t)MTOT * C_ * 2;   // 16 MB
  float*  alpha_acc = (float*)ws;                  ws += (size_t)MTOT * 4;
  float*  alpha     = (float*)ws;                  ws += (size_t)MTOT * 4;
  int*    cnt       = (int*)ws;                    ws += 256;

  // output carve-up: cs | cs_mask | loss_pen
  float* cs       = (float*)d_out;
  float* mask_out = cs + (size_t)B_ * CMAX * C_;
  float* loss     = mask_out + (size_t)B_ * CMAX * CMAX;

  k_init<<<MTOT / 256, 256, 0, stream>>>(alpha_acc, loss);
  k_cvt_bf16<<<(MTOT * C_ + 255) / 256, 256, 0, stream>>>(hs_pad, Xbf, MTOT * C_);
  k_cvt_bf16<<<(512 * 256 + 255) / 256, 256, 0, stream>>>(pw1_w, W1bf, 512 * 256);
  k_cvt_bf16<<<(256 * 256 + 255) / 256, 256, 0, stream>>>(pw2_w, W2bf, 256 * 256);

  k_gemm1_glu<<<dim3(4, 512), 256, 0, stream>>>(Xbf, W1bf, pw1_b, Y1);
  k_dwconv<<<(MTOT * C_ + 255) / 256, 256, 0, stream>>>(Y1, dw_w, dw_b, bn1_g, bn1_b, bn1_m, bn1_v, Zbf);
  k_gemm2_fused<<<dim3(4, 512), 256, 0, stream>>>(Zbf, W2bf, pw2_b, bn2_g, bn2_b, bn2_m, bn2_v, lin_w, alpha_acc);
  k_alpha<<<MTOT / 256, 256, 0, stream>>>(alpha_acc, hs_mask, lin_b, alpha, loss);

  k_cif<<<B_, 256, 0, stream>>>(alpha, hs_pad, cs, cnt);

  const size_t msk = (size_t)B_ * CMAX * CMAX;
  k_mask<<<(unsigned)((msk + 255) / 256), 256, 0, stream>>>(cnt, mask_out);
}
